// NodeApplyModule_5214090297536
// MI455X (gfx1250) — matrix operand order repaired and checked by reading, not compiled, not run on hardware
//
#include <hip/hip_runtime.h>

// ---------------------------------------------------------------------------
// InstanceNorm(512) + Linear(512->512) + ReLU, fused for MI455X (gfx1250).
//
//  * Normalization in f32 (exact), activations/weights cast to f16.
//  * GEMM via v_wmma_f32_16x16x32_f16 (f32 accumulate), wave32.
//  * One pass over x: block = 64 rows x full 512 outputs, 16 waves.
//  * Wave tile 32(M) x 64(N): 8 WMMAs per 12 B128 loads (4 ds + 8 global).
// ---------------------------------------------------------------------------

typedef __attribute__((ext_vector_type(16))) _Float16 v16h;
typedef __attribute__((ext_vector_type(8)))  _Float16 h8;
typedef __attribute__((ext_vector_type(4)))  _Float16 h4;
typedef __attribute__((ext_vector_type(8)))  float    v8f;
typedef __attribute__((ext_vector_type(4)))  float    f4;

#define KDIM        512
#define NOUT        512
#define M_TILE      64
#define LDS_STRIDE  520   // 512 + 8 halves pad -> consecutive rows 4 banks apart
#define EPS         1e-5f

// ---- Kernel 0: W f32 -> f16 (512 KB in d_ws; stays hot in L2 afterwards) ----
__global__ void w_to_f16_kernel(const float* __restrict__ W,
                                _Float16* __restrict__ Wh) {
    int i = (blockIdx.x * blockDim.x + threadIdx.x) * 4;
    f4 w = *(const f4*)(W + i);
    h4 h;
    h.x = (_Float16)w.x; h.y = (_Float16)w.y;
    h.z = (_Float16)w.z; h.w = (_Float16)w.w;
    *(h4*)(Wh + i) = h;
}

// ---- Kernel 1: fused instance-norm + GEMM + bias + ReLU ----
__launch_bounds__(512, 1)
__global__ void norm_gemm_kernel(const float* __restrict__ x,
                                 const _Float16* __restrict__ Wh,
                                 const float* __restrict__ bias,
                                 float* __restrict__ out) {
    __shared__ _Float16 lds[M_TILE * LDS_STRIDE];   // 66,560 B of the 320 KB WGP LDS

    const int tid  = threadIdx.x;
    const int lane = tid & 31;                      // wave32
    const int wave = tid >> 5;                      // 0..15
    const long row0 = (long)blockIdx.x * M_TILE;

    // ================= Phase 1: per-row InstanceNorm -> LDS (f16) ==========
    // Wave w normalizes rows 4w .. 4w+3. Each lane covers 16 of 512 elements.
    #pragma unroll
    for (int r = 0; r < 4; ++r) {
        const int lr = wave * 4 + r;                // local row in tile
        const float* xr = x + (row0 + lr) * KDIM;
        f4 v[4];
        float s = 0.f, sq = 0.f;
        #pragma unroll
        for (int c = 0; c < 4; ++c) {
            v[c] = *(const f4*)(xr + c * 128 + lane * 4);   // coalesced b128
            s  += v[c].x + v[c].y + v[c].z + v[c].w;
            sq += v[c].x * v[c].x + v[c].y * v[c].y
                + v[c].z * v[c].z + v[c].w * v[c].w;
        }
        // wave32 all-reduce
        #pragma unroll
        for (int off = 16; off > 0; off >>= 1) {
            s  += __shfl_xor(s,  off, 32);
            sq += __shfl_xor(sq, off, 32);
        }
        const float mean = s  * (1.0f / 512.0f);
        const float var  = sq * (1.0f / 512.0f) - mean * mean;  // biased var
        const float rs   = rsqrtf(var + EPS);
        _Float16* dst = lds + lr * LDS_STRIDE;
        #pragma unroll
        for (int c = 0; c < 4; ++c) {
            h4 h;
            h.x = (_Float16)((v[c].x - mean) * rs);
            h.y = (_Float16)((v[c].y - mean) * rs);
            h.z = (_Float16)((v[c].z - mean) * rs);
            h.w = (_Float16)((v[c].w - mean) * rs);
            *(h4*)(dst + c * 128 + lane * 4) = h;   // ds_store_b64
        }
    }
    __syncthreads();

    // ================= Phase 2: WMMA GEMM =================================
    // 16 waves = 2 (M groups of 32 rows) x 8 (N groups of 64 cols).
    // Per wave: 2x4 C tiles (16x16 f32) = 64 accumulator VGPRs/lane.
    // Each A frag feeds 4 WMMAs, each B frag feeds 2 WMMAs.
    const int mg  = wave & 1;                       // 0..1 -> rows mg*32
    const int ng  = wave >> 1;                      // 0..7 -> cols ng*64
    const int l16 = lane & 15;
    const bool hi = (lane >= 16);

    v8f acc[2][4] = {};

    // A fragment addressing (dense 16-bit A 16x32 ISA layout):
    //   lane<16  : K {k0..k0+7 , k0+16..k0+23}
    //   lane>=16 : K {k0+8..k0+15, k0+24..k0+31}
    const _Float16* aBase = lds + (mg * 32 + l16) * LDS_STRIDE + (hi ? 8 : 0);
    // B fragment addressing (16-bit B 32x16, same K order as A):
    //   lane<16 : K {k0..k0+7, k0+16..k0+23} of W-row (output) o;  lane>=16 : K {k0+8..k0+15, k0+24..k0+31}
    const _Float16* bBase = Wh + (long)(ng * 64 + l16) * KDIM + (hi ? 8 : 0);

    for (int k0 = 0; k0 < KDIM; k0 += 32) {
        // two A fragments (rows mg*32 .. +15 and +16 .. +31)
        v16h a[2];
        #pragma unroll
        for (int tm = 0; tm < 2; ++tm) {
            const _Float16* ap = aBase + tm * 16 * LDS_STRIDE + k0;
            h8 alo = *(const h8*)(ap);              // ds_load_b128
            h8 ahi = *(const h8*)(ap + 16);         // ds_load_b128
            a[tm] = __builtin_shufflevector(alo, ahi,
                    0,1,2,3,4,5,6,7,8,9,10,11,12,13,14,15);
        }
        #pragma unroll
        for (int tn = 0; tn < 4; ++tn) {
            const _Float16* bp = bBase + (long)tn * 16 * KDIM + k0;
            h8 blo = *(const h8*)(bp);              // global_load_b128 (L2-hot W)
            h8 bhi = *(const h8*)(bp + 16);         // global_load_b128
            v16h b = __builtin_shufflevector(blo, bhi,
                     0,1,2,3,4,5,6,7,8,9,10,11,12,13,14,15);
            #pragma unroll
            for (int tm = 0; tm < 2; ++tm) {
                acc[tm][tn] = __builtin_amdgcn_wmma_f32_16x16x32_f16(
                                  /*neg_a=*/false, a[tm], /*neg_b=*/false, b,
                                  /*c_mod=*/(short)0, acc[tm][tn],
                                  /*reuse_a=*/false, /*reuse_b=*/false);
            }
        }
    }

    // ================= Epilogue: bias + ReLU + store ======================
    // C layout: VGPR i -> M = i + 8*(lane>=16), N = lane%16.
    const int mtop = hi ? 8 : 0;
    #pragma unroll
    for (int tn = 0; tn < 4; ++tn) {
        const int o  = ng * 64 + tn * 16 + l16;
        const float bb = bias[o];
        #pragma unroll
        for (int tm = 0; tm < 2; ++tm) {
            float* op = out + (row0 + mg * 32 + tm * 16 + mtop) * NOUT + o;
            #pragma unroll
            for (int i = 0; i < 8; ++i) {
                float v = acc[tm][tn][i] + bb;
                op[(long)i * NOUT] = v > 0.f ? v : 0.f;   // half-wave 64B rows
            }
        }
    }
}

// ---------------------------------------------------------------------------
extern "C" void kernel_launch(void* const* d_in, const int* in_sizes, int n_in,
                              void* d_out, int out_size, void* d_ws, size_t ws_size,
                              hipStream_t stream) {
    const float* x = (const float*)d_in[0];     // [N, 512]
    const float* W = (const float*)d_in[1];     // [512, 512] row-major (o, i)
    const float* b = (const float*)d_in[2];     // [512]
    float* out     = (float*)d_out;             // [N, 512]
    _Float16* Wh   = (_Float16*)d_ws;           // 512 KB scratch

    const int nrows = in_sizes[0] / KDIM;       // 200000 (divisible by 64)

    // W -> f16 : 262144 elems, 4 per thread
    w_to_f16_kernel<<<(KDIM * NOUT) / (256 * 4), 256, 0, stream>>>(W, Wh);

    // fused norm + GEMM: 64 rows per block, 512 threads (16 waves)
    norm_gemm_kernel<<<nrows / M_TILE, 512, 0, stream>>>(x, Wh, b, out);
}
